// PrefixSumCounts_1125281431611
// MI455X (gfx1250) — matrix operand order repaired and checked by reading, not compiled, not run on hardware
//
#include <hip/hip_runtime.h>

typedef __attribute__((ext_vector_type(16))) _Float16 v16h;
typedef __attribute__((ext_vector_type(8)))  float    v8f;

#define BB 8
#define TT 4096
#define VV 2048
#define NSEG 8
#define SEGLEN (TT / NSEG)    // 512 tokens per wave
#define NTILE (SEGLEN / 16)   // 32 WMMA tiles per wave

// select element i (0..7) of a v8f without dynamic indexing (v_cndmask chain)
__device__ __forceinline__ float sel8(v8f v, int i) {
    float r = v[0];
    r = (i == 1) ? v[1] : r;
    r = (i == 2) ? v[2] : r;
    r = (i == 3) ? v[3] : r;
    r = (i == 4) ? v[4] : r;
    r = (i == 5) ? v[5] : r;
    r = (i == 6) ? v[6] : r;
    r = (i == 7) ? v[7] : r;
    return r;
}

__global__ __launch_bounds__(256) void runcount_wmma_kernel(
    const int* __restrict__ x, float* __restrict__ out) {
    // per-segment running histograms: 8 x 2048 x u32 = 64 KB LDS
    __shared__ unsigned h[NSEG][VV];

    const int tid  = threadIdx.x;
    const int lane = tid & 31;
    const int seg  = tid >> 5;        // wave id = segment id
    const int row  = blockIdx.x;
    const int M    = lane & 15;       // matrix row owned by this lane
    const bool hi  = lane >= 16;      // high lane-half

    const int*  xr   = x + row * TT;
    float*      outr = out + row * TT;

    // ---- zero histograms ----
    for (int i = tid; i < NSEG * VV; i += 256) ((unsigned*)h)[i] = 0u;
    __syncthreads();

    // ---- Phase A: per-segment histogram (16 iters of 32 lanes) ----
    {
        const int base = seg * SEGLEN;
        for (int i = 0; i < SEGLEN; i += 32) {
            int t = xr[base + i + lane];
            atomicAdd(&h[seg][t], 1u);
        }
    }
    __syncthreads();

    // ---- exclusive prefix across segments, per vocab bin ----
    for (int v = tid; v < VV; v += 256) {
        unsigned run = 0;
#pragma unroll
        for (int s = 0; s < NSEG; ++s) {
            unsigned c = h[s][v];
            h[s][v] = run;
            run += c;
        }
    }
    __syncthreads();

    // ---- constant A2 = lower-triangular ones (incl. diagonal), A-layout ----
    // A 16x32 f16: lane M = lane&15; k order used here is 16h+s: lanes0-15 elem s -> K = s; lanes16-31 (K>=16) zero
    v16h a2 = {};
#pragma unroll
    for (int v = 0; v < 8; ++v) {
        int k0 = 2 * v;
        a2[2 * v]     = (_Float16)((!hi && k0     <= M) ? 1.0f : 0.0f);
        a2[2 * v + 1] = (_Float16)((!hi && k0 + 1 <= M) ? 1.0f : 0.0f);
    }
    const v8f zero8 = {};

    // ---- Phase B: scan segment in 16-token tiles, ranks via 2x WMMA ----
    for (int j = 0; j < NTILE; ++j) {
        const int tb = seg * SEGLEN + j * 16;
        const int tm = xr[tb + M];          // lane k and lane k+16 both hold tok_k

        // broadcast tile tokens to all lanes
        int t16[16];
#pragma unroll
        for (int k = 0; k < 16; ++k) t16[k] = __shfl(tm, k);

        // A1[i][k] = [tok_i == tok_k]  (A-layout, K>=16 zero)
        v16h a1 = {};
#pragma unroll
        for (int v = 0; v < 8; ++v) {
            int e0 = (!hi) && (tm == t16[2 * v]);
            int e1 = (!hi) && (tm == t16[2 * v + 1]);
            a1[2 * v]     = (_Float16)(float)e0;
            a1[2 * v + 1] = (_Float16)(float)e1;
        }
        // B1[k][j] = [tok_j == tok_k]  (B operand, same k = 16h+s order as A: lanes0-15 elem s -> K=s; lanes16-31 -> 0)
        v16h b1 = {};
#pragma unroll
        for (int v = 0; v < 8; ++v) {
            int e0 = (!hi) && (tm == t16[2 * v]);
            int e1 = (!hi) && (tm == t16[2 * v + 1]);
            b1[2 * v]     = (_Float16)(float)e0;
            b1[2 * v + 1] = (_Float16)(float)e1;
        }

        // P[i][j] = m_i if tok_i==tok_j else 0   (16x16 f32, C-layout)
        v8f P = __builtin_amdgcn_wmma_f32_16x16x32_f16(
            false, a1, false, b1, (short)0, zero8, false, false);

        // B2 = P as f16 B-operand: lanes0-15, VGPR v holds rows 2v,2v+1 at col=lane
        v16h b2 = {};
#pragma unroll
        for (int v = 0; v < 4; ++v) {           // rows 0..7: own lane, C-VGPR = row
            b2[2 * v]     = (_Float16)(hi ? 0.0f : P[2 * v]);
            b2[2 * v + 1] = (_Float16)(hi ? 0.0f : P[2 * v + 1]);
        }
#pragma unroll
        for (int v = 4; v < 8; ++v) {           // rows 8..15 live in lanes16-31 of C
            float r0 = __shfl(P[2 * v - 8], lane + 16);
            float r1 = __shfl(P[2 * v - 7], lane + 16);
            b2[2 * v]     = (_Float16)(hi ? 0.0f : r0);
            b2[2 * v + 1] = (_Float16)(hi ? 0.0f : r1);
        }

        // Q[i][j] = sum_{k<=i} P[k][j]; diag: Q[i][i] = m_i * rank_i (P symmetric)
        v8f Q = __builtin_amdgcn_wmma_f32_16x16x32_f16(
            false, a2, false, b2, (short)0, zero8, false, false);

        // extract diagonals: (i,i) at [VGPR i, lane i] for i<8, [VGPR i-8, lane i+16] for i>=8
        int   i3   = lane & 7;
        float pSel = sel8(P, i3);
        float qSel = sel8(Q, i3);
        int   src  = (lane & 8) ? (lane + 16) : lane;
        float Pd   = __shfl(pSel, src);   // multiplicity m_i
        float Qd   = __shfl(qSel, src);   // m_i * rank_i
        float mf   = (Pd > 0.0f) ? Pd : 1.0f;
        int   ci   = (int)(Qd / mf + 0.5f);   // inclusive rank within tile
        int   mi   = (int)(Pd + 0.5f);

        // cross-tile base from running histogram (read BEFORE this tile's update;
        // LDS ops from one wave are in-order, segment region is wave-private)
        unsigned baseCnt = h[seg][tm];

        if (lane < 16) {
            outr[tb + M] = (float)(baseCnt + (unsigned)ci);
            if (ci == 1)  // first occurrence in tile updates running histogram
                atomicAdd(&h[seg][tm], (unsigned)mi);
        }
        __threadfence_block();  // keep update->next-tile-read ordered for the compiler
    }
}

extern "C" void kernel_launch(void* const* d_in, const int* in_sizes, int n_in,
                              void* d_out, int out_size, void* d_ws, size_t ws_size,
                              hipStream_t stream) {
    const int* x = (const int*)d_in[0];   // token ids [8, 4096], int32
    // d_in[1] (one_hot, 256 MB fp32) intentionally unused: output depends only on x.
    float* out = (float*)d_out;           // [8, 4096, 1] fp32
    (void)in_sizes; (void)n_in; (void)out_size; (void)d_ws; (void)ws_size;
    runcount_wmma_kernel<<<BB, 256, 0, stream>>>(x, out);
}
